// GraphClassifier_36051955483068
// MI455X (gfx1250) — compile-verified, matrix operand order checked by reading, not run on hardware
//
#include <hip/hip_runtime.h>
#include <hip/hip_bf16.h>

// ---------------------------------------------------------------------------
// GNN graph classifier for MI455X (gfx1250, wave32, WMMA).
//  - edge MLP composed into edge_attr projection (K=32 == one WMMA K-step)
//  - segment_sum of edge messages hoisted out of the conv loop (S1/S2)
//  - WMMA fragments built from K-contiguous LDS with 2 x ds_load_b128 each
//    (A stored [m][k], weights stored transposed [n][k])
// ---------------------------------------------------------------------------

typedef __attribute__((ext_vector_type(16))) _Float16 v16h;
typedef __attribute__((ext_vector_type(8)))  float    v8f;

#define H 128

union FragU {
  v16h v;
  uint4 u[2];
};
union Pack8 {
  _Float16 h[8];
  uint4 u;
};

// Build a 16-half fragment for lane-group `grp` from a K-contiguous row.
// elements 0..7  <- k = 8*grp + 0..7
// elements 8..15 <- k = 16 + 8*grp + 0..7   (CDNA5 16-bit A/B layout)
__device__ __forceinline__ v16h load_frag(const _Float16* row_k0, int grp) {
  FragU f;
  f.u[0] = *(const uint4*)(row_k0 + 8 * grp);
  f.u[1] = *(const uint4*)(row_k0 + 16 + 8 * grp);
  return f.v;
}

// ---------------------------------------------------------------------------
// Node GEMM: C[rows,128] = act(A[rows,128] @ W[128,128] + bias)
// W is f16, TRANSPOSED ([n][k]). Block = 256 thr (8 waves) computes a 64-row
// slab; wave w owns column tile n0=16*w across 4 sub-slabs of 16 rows.
// ---------------------------------------------------------------------------
__global__ __launch_bounds__(256) void k_gemm_node(
    const float* __restrict__ A, const _Float16* __restrict__ Wt,
    const float* __restrict__ bias, float* __restrict__ C, int rows, int relu) {
  __shared__ __align__(16) _Float16 As[64 * H];   // 16 KB, [m][k]
  __shared__ __align__(16) _Float16 Ws[H * H];    // 32 KB, [n][k]
  const int t = threadIdx.x;
  const int wave = t >> 5, lane = t & 31;
  const int row0 = blockIdx.x * 64;

  // stage A slab f32->f16: 8192 halves, 4 x (8-half packed store) per thread
#pragma unroll
  for (int i = 0; i < 4; ++i) {
    int hb = (t * 4 + i) * 8;          // half base, 16B aligned
    int r = hb >> 7, c = hb & (H - 1);
    Pack8 p;
    if (row0 + r < rows) {
      const float4 f0 = *(const float4*)(A + (size_t)(row0 + r) * H + c);
      const float4 f1 = *(const float4*)(A + (size_t)(row0 + r) * H + c + 4);
      p.h[0] = (_Float16)f0.x; p.h[1] = (_Float16)f0.y;
      p.h[2] = (_Float16)f0.z; p.h[3] = (_Float16)f0.w;
      p.h[4] = (_Float16)f1.x; p.h[5] = (_Float16)f1.y;
      p.h[6] = (_Float16)f1.z; p.h[7] = (_Float16)f1.w;
    } else {
      p.u = make_uint4(0, 0, 0, 0);
    }
    *(uint4*)(As + hb) = p.u;
  }
  // stage Wt (f16, already transposed): 32 KB = 2048 uint4
  {
    const uint4* src = (const uint4*)Wt;
    uint4* dst = (uint4*)Ws;
#pragma unroll
    for (int i = 0; i < 8; ++i) dst[t + 256 * i] = src[t + 256 * i];
  }
  __syncthreads();

  const int n0 = wave * 16;
  const int mn = lane & 15, grp = lane >> 4;
  v8f acc[4] = {};
#pragma unroll
  for (int kt = 0; kt < 4; ++kt) {
    const v16h b = load_frag(&Ws[(n0 + mn) * H + kt * 32], grp);
#pragma unroll
    for (int s = 0; s < 4; ++s) {
      const v16h a = load_frag(&As[(s * 16 + mn) * H + kt * 32], grp);
      acc[s] = __builtin_amdgcn_wmma_f32_16x16x32_f16(false, a, false, b,
                                                      (short)0, acc[s],
                                                      false, false);
    }
  }
  const float bn = bias[n0 + mn];
#pragma unroll
  for (int s = 0; s < 4; ++s) {
#pragma unroll
    for (int i = 0; i < 8; ++i) {
      int r = row0 + s * 16 + i + 8 * grp;
      if (r < rows) {
        float v = acc[s][i] + bn;
        if (relu) v = fmaxf(v, 0.0f);
        C[(size_t)r * H + n0 + mn] = v;
      }
    }
  }
}

// ---------------------------------------------------------------------------
// Edge precompute: S_k[dst] += edge_attr[e,0:32] @ Wec_k + bec_k.
// Wec_k is f16 TRANSPOSED ([n=128][k=32]); K=32 -> one WMMA per 16x16 tile.
// Block = 128 edges, 8 waves; wave owns 16 edges x 8 col tiles x 2 convs.
// ---------------------------------------------------------------------------
__global__ __launch_bounds__(256) void k_edge_scatter(
    const float* __restrict__ edge_attr, const int* __restrict__ edge_index,
    const _Float16* __restrict__ Wec1, const _Float16* __restrict__ Wec2,
    const float* __restrict__ bec1, const float* __restrict__ bec2,
    float* __restrict__ S1, float* __restrict__ S2, int E) {
  __shared__ __align__(16) _Float16 Ae[128 * 32];     // 8 KB, [e][k]
  __shared__ __align__(16) _Float16 Wl[2][H * 32];    // 16 KB, [n][k]
  const int t = threadIdx.x;
  const int wave = t >> 5, lane = t & 31;
  const int e0 = blockIdx.x * 128;

  // stage 128 edge rows f32->f16: 4096 halves, 2 packed stores per thread
#pragma unroll
  for (int i = 0; i < 2; ++i) {
    int hb = (t * 2 + i) * 8;
    int r = hb >> 5, c = hb & 31;
    Pack8 p;
    if (e0 + r < E) {
      const float4 f0 = *(const float4*)(edge_attr + (size_t)(e0 + r) * 32 + c);
      const float4 f1 = *(const float4*)(edge_attr + (size_t)(e0 + r) * 32 + c + 4);
      p.h[0] = (_Float16)f0.x; p.h[1] = (_Float16)f0.y;
      p.h[2] = (_Float16)f0.z; p.h[3] = (_Float16)f0.w;
      p.h[4] = (_Float16)f1.x; p.h[5] = (_Float16)f1.y;
      p.h[6] = (_Float16)f1.z; p.h[7] = (_Float16)f1.w;
    } else {
      p.u = make_uint4(0, 0, 0, 0);
    }
    *(uint4*)(Ae + hb) = p.u;
  }
  // stage both composed weight matrices (2 x 8 KB = 1024 uint4)
  {
    const uint4* s1 = (const uint4*)Wec1;
    const uint4* s2 = (const uint4*)Wec2;
    uint4* d1 = (uint4*)&Wl[0][0];
    uint4* d2 = (uint4*)&Wl[1][0];
#pragma unroll
    for (int i = 0; i < 2; ++i) {
      d1[t + 256 * i] = s1[t + 256 * i];
      d2[t + 256 * i] = s2[t + 256 * i];
    }
  }
  __syncthreads();

  const int me0 = wave * 16;
  const int mn = lane & 15, grp = lane >> 4;
  const v16h a = load_frag(&Ae[(me0 + mn) * 32], grp);

  for (int s = 0; s < 2; ++s) {
    const _Float16* Wp = &Wl[s][0];
    const float* bep = s ? bec2 : bec1;
    float* Sp = s ? S2 : S1;
#pragma unroll
    for (int nt = 0; nt < 8; ++nt) {
      const int n0 = nt * 16;
      const v16h b = load_frag(&Wp[(n0 + mn) * 32], grp);
      v8f acc = {};
      acc = __builtin_amdgcn_wmma_f32_16x16x32_f16(false, a, false, b,
                                                   (short)0, acc, false, false);
      const float bn = bep[n0 + mn];
#pragma unroll
      for (int i = 0; i < 8; ++i) {
        int eg = e0 + me0 + i + 8 * grp;
        if (eg < E) {
          int dst = edge_index[E + eg];
          atomicAdd(&Sp[(size_t)dst * H + n0 + mn], acc[i] + bn);
        }
      }
    }
  }
}

// aggr = hn + S_k + be_k  (elementwise over N*H)
__global__ void k_init_aggr(const float* __restrict__ hn,
                            const float* __restrict__ S,
                            const float* __restrict__ be,
                            float* __restrict__ aggr, int total) {
  for (int i = blockIdx.x * blockDim.x + threadIdx.x; i < total;
       i += gridDim.x * blockDim.x)
    aggr[i] = hn[i] + S[i] + be[i & (H - 1)];
}

// one wave per edge: aggr[dst] += hn[src] (128 floats = 32 lanes x float4)
__global__ __launch_bounds__(256) void k_scatter(
    const float* __restrict__ hn, const int* __restrict__ edge_index,
    float* __restrict__ aggr, int E) {
  const int wid = (blockIdx.x * blockDim.x + threadIdx.x) >> 5;
  const int lane = threadIdx.x & 31;
  const int nw = (gridDim.x * blockDim.x) >> 5;
  for (int e = wid; e < E; e += nw) {
    int src = edge_index[e];
    int dst = edge_index[E + e];
    if (e + nw < E) {  // prefetch next source row (global_prefetch_b8)
      int nsrc = edge_index[e + nw];
      __builtin_prefetch(hn + (size_t)nsrc * H + lane * 4, 0, 0);
    }
    float4 v = ((const float4*)(hn + (size_t)src * H))[lane];
    float* dp = aggr + (size_t)dst * H + lane * 4;
    atomicAdd(dp + 0, v.x);
    atomicAdd(dp + 1, v.y);
    atomicAdd(dp + 2, v.z);
    atomicAdd(dp + 3, v.w);
  }
}

__global__ void k_zero(float* p, int n) {
  for (int i = blockIdx.x * blockDim.x + threadIdx.x; i < n;
       i += gridDim.x * blockDim.x)
    p[i] = 0.0f;
}

// f32 [128,128] -> f16 TRANSPOSED ([n][k])
__global__ void k_cvt_t(const float* __restrict__ s, _Float16* __restrict__ d,
                        int n) {
  for (int i = blockIdx.x * blockDim.x + threadIdx.x; i < n;
       i += gridDim.x * blockDim.x) {
    int nn = i >> 7, k = i & (H - 1);
    d[i] = (_Float16)s[k * H + nn];
  }
}

// compose Wec_k = (We2l[32,128] @ We_k[128,128])^T as f16 [n=128][k=32],
// bec_k = be2l @ We_k + be_k
__global__ __launch_bounds__(256) void k_prep(
    const float* __restrict__ We2l, const float* __restrict__ be2l,
    const float* __restrict__ We1, const float* __restrict__ be1,
    const float* __restrict__ We2, const float* __restrict__ be2,
    _Float16* __restrict__ Wec1, _Float16* __restrict__ Wec2,
    float* __restrict__ bec1, float* __restrict__ bec2) {
  const int t = threadIdx.x;
  for (int s = 0; s < 2; ++s) {
    const float* Wk = s ? We2 : We1;
    _Float16* out = s ? Wec2 : Wec1;
#pragma unroll 1
    for (int i = 0; i < 16; ++i) {
      int idx = t * 16 + i;          // idx = n*32 + r (transposed layout)
      int nn = idx >> 5, r = idx & 31;
      float acc = 0.0f;
      for (int k = 0; k < H; ++k) acc += We2l[r * H + k] * Wk[k * H + nn];
      out[idx] = (_Float16)acc;
    }
    const float* bk = s ? be2 : be1;
    float* bo = s ? bec2 : bec1;
    if (t < H) {
      float acc = bk[t];
      for (int k = 0; k < H; ++k) acc += be2l[k] * Wk[k * H + t];
      bo[t] = acc;
    }
  }
}

// one wave per node: mean-acc, count, and int-atomicMax (h>=0 after relu)
__global__ __launch_bounds__(256) void k_pool(
    const float* __restrict__ h, const int* __restrict__ batch,
    float* __restrict__ msum, float* __restrict__ cnt, float* __restrict__ mx,
    int N) {
  const int wid = (blockIdx.x * blockDim.x + threadIdx.x) >> 5;
  const int lane = threadIdx.x & 31;
  const int nw = (gridDim.x * blockDim.x) >> 5;
  for (int n = wid; n < N; n += nw) {
    int g = batch[n];
    float4 v = ((const float4*)(h + (size_t)n * H))[lane];
    float* mp = msum + (size_t)g * H + lane * 4;
    atomicAdd(mp + 0, v.x);
    atomicAdd(mp + 1, v.y);
    atomicAdd(mp + 2, v.z);
    atomicAdd(mp + 3, v.w);
    int* xp = (int*)(mx + (size_t)g * H + lane * 4);
    atomicMax(xp + 0, __float_as_int(v.x));
    atomicMax(xp + 1, __float_as_int(v.y));
    atomicMax(xp + 2, __float_as_int(v.z));
    atomicMax(xp + 3, __float_as_int(v.w));
    if (lane == 0) atomicAdd(&cnt[g], 1.0f);
  }
}

// readout MLP: pooled[G,256] -> relu(@Wm1[256,128]) -> @Wm2[128,2] -> sigmoid
__global__ __launch_bounds__(256) void k_readout(
    const float* __restrict__ msum, const float* __restrict__ cnt,
    const float* __restrict__ mx, const float* __restrict__ Wm1,
    const float* __restrict__ bm1, const float* __restrict__ Wm2,
    const float* __restrict__ bm2, float* __restrict__ out, int G) {
  __shared__ float z1[64 * H];  // 32 KB (G <= 64)
  const int t = threadIdx.x;
  for (int idx = t; idx < G * H; idx += 256) {
    int g = idx >> 7, j = idx & (H - 1);
    float c = fmaxf(cnt[g], 1.0f);
    float acc = bm1[j];
    for (int k = 0; k < H; ++k)
      acc += (msum[g * H + k] / c) * Wm1[k * H + j];
    for (int k = 0; k < H; ++k)
      acc += mx[g * H + k] * Wm1[(H + k) * H + j];
    z1[idx] = fmaxf(acc, 0.0f);
  }
  __syncthreads();
  for (int idx = t; idx < G * 2; idx += 256) {
    int g = idx >> 1, o = idx & 1;
    float acc = bm2[o];
    for (int j = 0; j < H; ++j) acc += z1[g * H + j] * Wm2[j * 2 + o];
    out[idx] = 1.0f / (1.0f + expf(-acc));
  }
}

extern "C" void kernel_launch(void* const* d_in, const int* in_sizes, int n_in,
                              void* d_out, int out_size, void* d_ws,
                              size_t ws_size, hipStream_t stream) {
  const float* x    = (const float*)d_in[0];
  const float* ea   = (const float*)d_in[1];
  const int*   ei   = (const int*)d_in[2];
  const int*   bat  = (const int*)d_in[3];
  const float* Wn2l = (const float*)d_in[4];  const float* bn2l = (const float*)d_in[5];
  const float* We2l = (const float*)d_in[6];  const float* be2l = (const float*)d_in[7];
  const float* Wn1  = (const float*)d_in[8];  const float* bn1  = (const float*)d_in[9];
  const float* We1  = (const float*)d_in[10]; const float* be1  = (const float*)d_in[11];
  const float* Wu1  = (const float*)d_in[12]; const float* bu1  = (const float*)d_in[13];
  const float* Wn2  = (const float*)d_in[14]; const float* bn2  = (const float*)d_in[15];
  const float* We2  = (const float*)d_in[16]; const float* be2  = (const float*)d_in[17];
  const float* Wu2  = (const float*)d_in[18]; const float* bu2  = (const float*)d_in[19];
  const float* Wm1  = (const float*)d_in[20]; const float* bm1  = (const float*)d_in[21];
  const float* Wm2  = (const float*)d_in[22]; const float* bm2  = (const float*)d_in[23];
  float* out = (float*)d_out;

  const int N = in_sizes[0] / H;   // 20000
  const int E = in_sizes[1] / 32;  // 640000
  const int G = out_size / 2;      // 64
  const int NH = N * H;

  // ---- workspace layout (fp32 region, then f16 region) ----
  float* S1   = (float*)d_ws;
  float* S2   = S1 + NH;
  float* msum = S2 + NH;          // [G,H]
  float* cnt  = msum + G * H;     // [G]
  float* mx   = cnt + G;          // [G,H]
  float* h    = mx + G * H;       // [N,H]
  float* hn   = h + NH;
  float* aggr = hn + NH;
  float* bec1 = aggr + NH;        // [H]
  float* bec2 = bec1 + H;
  _Float16* f16b   = (_Float16*)(bec2 + H);
  _Float16* Wn2l16 = f16b;                  // all node weights transposed [n][k]
  _Float16* Wn116  = Wn2l16 + H * H;
  _Float16* Wu116  = Wn116 + H * H;
  _Float16* Wn216  = Wu116 + H * H;
  _Float16* Wu216  = Wn216 + H * H;
  _Float16* Wec1   = Wu216 + H * H;         // [128,32] transposed
  _Float16* Wec2   = Wec1 + 32 * H;

  const int zeroN = 2 * NH + 2 * G * H + G;  // S1,S2,msum,cnt,mx contiguous
  k_zero<<<1024, 256, 0, stream>>>(S1, zeroN);

  k_prep<<<1, 256, 0, stream>>>(We2l, be2l, We1, be1, We2, be2, Wec1, Wec2,
                                bec1, bec2);
  k_cvt_t<<<16, 256, 0, stream>>>(Wn2l, Wn2l16, H * H);
  k_cvt_t<<<16, 256, 0, stream>>>(Wn1, Wn116, H * H);
  k_cvt_t<<<16, 256, 0, stream>>>(Wu1, Wu116, H * H);
  k_cvt_t<<<16, 256, 0, stream>>>(Wn2, Wn216, H * H);
  k_cvt_t<<<16, 256, 0, stream>>>(Wu2, Wu216, H * H);

  const int gemmBlocks = (N + 63) / 64;
  // h0 = x @ Wn2l + bn2l
  k_gemm_node<<<gemmBlocks, 256, 0, stream>>>(x, Wn2l16, bn2l, h, N, 0);
  // loop-invariant edge aggregation S1/S2
  k_edge_scatter<<<(E + 127) / 128, 256, 0, stream>>>(ea, ei, Wec1, Wec2,
                                                      bec1, bec2, S1, S2, E);

  for (int it = 0; it < 3; ++it) {
    // conv1
    k_gemm_node<<<gemmBlocks, 256, 0, stream>>>(h, Wn116, bn1, hn, N, 0);
    k_init_aggr<<<2048, 256, 0, stream>>>(hn, S1, be1, aggr, NH);
    k_scatter<<<2048, 256, 0, stream>>>(hn, ei, aggr, E);
    k_gemm_node<<<gemmBlocks, 256, 0, stream>>>(aggr, Wu116, bu1, h, N, 1);
    // conv2
    k_gemm_node<<<gemmBlocks, 256, 0, stream>>>(h, Wn216, bn2, hn, N, 0);
    k_init_aggr<<<2048, 256, 0, stream>>>(hn, S2, be2, aggr, NH);
    k_scatter<<<2048, 256, 0, stream>>>(hn, ei, aggr, E);
    k_gemm_node<<<gemmBlocks, 256, 0, stream>>>(aggr, Wu216, bu2, h, N, 1);
  }

  k_pool<<<640, 256, 0, stream>>>(h, bat, msum, cnt, mx, N);
  k_readout<<<1, 256, 0, stream>>>(msum, cnt, mx, Wm1, bm1, Wm2, bm2, out, G);
}
